// Tanh_TCN_62165356642409
// MI455X (gfx1250) — matrix operand order repaired and checked by reading, not compiled, not run on hardware
//
#include <hip/hip_runtime.h>
#include <hip/hip_bf16.h>
#include <math.h>

typedef _Float16 f16;
typedef __attribute__((ext_vector_type(16))) _Float16 v16h;
typedef __attribute__((ext_vector_type(8)))  float    v8f;

#define PI_F 3.14159265358979f

#define Bsz 32
#define Cch 32
#define Tn  1024
#define NFb 33
#define NWIN 961
#define RT_PER_B 61          // 61*16 = 976 padded rows per batch element
#define MPAD_PER_B 976
#define ROWTILES (Bsz*RT_PER_B)   // 1952 row tiles of 16

// ---- WMMA f16 16x16x32 fragment index helpers (per CDNA5 ISA 7.12.2) ----
// A (16x32, 16-bit): lane = m + 16*bit3(k); half-slot = (k&7) | ((k&16)>>1)
__device__ __forceinline__ int a_lane(int m, int kl){ return m + (((kl>>3)&1)<<4); }
__device__ __forceinline__ int a_h(int kl){ return (kl&7) | ((kl&16)>>1); }

// ---- repack W[N][K] (f32 row-major) into B-fragment layout:
// B element (k,n): lane = (n&15) + 16*bit3(k_local), half = (k_local&7) | ((k_local&16)>>1)
__global__ void k_convW(const float* __restrict__ W, f16* __restrict__ Bf, int N, int K)
{
  int idx = blockIdx.x*blockDim.x + threadIdx.x;
  if (idx >= N*K) return;
  int n = idx / K, k = idx - n*K;
  int kt = k>>5, kl = k&31, ct = n>>4, nl = n&15;
  int lane = nl + (((kl>>3)&1)<<4);
  int h = (kl&7) | ((kl&16)>>1);
  int CT = N >> 4;
  Bf[(size_t)((kt*CT + ct)*32 + lane)*16 + h] = (f16)W[idx];
}

// ---- build 64x80 DFT basis (cols 2n=cos, 2n+1=-sin, 66..79 zero) in B-frag layout ----
__global__ void k_dftbasis(f16* __restrict__ Bf)
{
  int idx = blockIdx.x*blockDim.x + threadIdx.x;
  if (idx >= 64*80) return;
  int k = idx / 80, j = idx - k*80;
  float v = 0.f;
  if (j < 2*NFb) {
    int n = j >> 1;
    float ang = 2.f*PI_F*(float)(k*n)*(1.f/64.f);
    v = (j & 1) ? -__sinf(ang) : __cosf(ang);
  }
  int kt = k>>5, kl = k&31, ct = j>>4, jl = j&15;
  int lane = jl + (((kl>>3)&1)<<4);
  int h = (kl&7) | ((kl&16)>>1);
  Bf[(size_t)((kt*5 + ct)*32 + lane)*16 + h] = (f16)v;
}

// ---- softmax stats over agg[0..960] + zero the output accumulator ----
__global__ __launch_bounds__(1024) void k_stats(const float* __restrict__ agg,
                                                float* __restrict__ stats,
                                                float* __restrict__ out)
{
  __shared__ float red[1024];
  int tid = threadIdx.x;
  if (tid < Bsz) out[tid] = 0.f;
  float v = (tid < NWIN) ? agg[tid] : -1e30f;
  red[tid] = v; __syncthreads();
  for (int s = 512; s > 0; s >>= 1) {
    if (tid < s) red[tid] = fmaxf(red[tid], red[tid+s]);
    __syncthreads();
  }
  float maxa = red[0]; __syncthreads();
  red[tid] = (tid < NWIN) ? expf(agg[tid]-maxa) : 0.f; __syncthreads();
  for (int s = 512; s > 0; s >>= 1) {
    if (tid < s) red[tid] += red[tid+s];
    __syncthreads();
  }
  if (tid == 0) { stats[0] = maxa; stats[1] = red[0]; }
}

// ---- fused WMMA DFT + feature nonlinearity + WMMA GEMM (feat @ Wp^T) ----
// One WG = 4 waves handles 16 windows (one b) x all 256 output cols.
// K = 2112 processed as two chunks of 16 channels = 33 aligned K-tiles each.
// DFT is done on the matrix units: windows(16x64) @ basis(64x80) per channel.
__global__ __launch_bounds__(128) void k_dft_gemm(
    const float* __restrict__ x, const f16* __restrict__ WdftB,
    const f16* __restrict__ WpB, const float* __restrict__ bp,
    f16* __restrict__ projA)
{
  __shared__ __align__(32) f16 ldsXh[Cch][80];     // 5 KB
  __shared__ float ldsF[4][16][80];                // 20 KB  per-wave DFT scratch
  __shared__ __align__(32) f16 ldsA[33*512];       // 33.8 KB A fragments for one chunk

  const int tid  = threadIdx.x;
  const int lane = tid & 31;
  const int wave = tid >> 5;
  const int rt   = blockIdx.x;            // global row tile
  const int b    = rt / RT_PER_B;
  const int rtl  = rt - b*RT_PER_B;
  const int w0   = rtl*16;

  // stage x window block into LDS as f16: 32 channels x 79 samples (clamped)
  for (int i = tid; i < Cch*79; i += 128) {
    int c = i / 79, j = i - c*79;
    int t = w0 + j; if (t > Tn-1) t = Tn-1;
    ldsXh[c][j] = (f16)x[(size_t)(b*Cch + c)*Tn + t];
  }

  v8f acc[4];
  #pragma unroll
  for (int i = 0; i < 4; ++i) acc[i] = (v8f)(0.f);

  const int m_l = lane & 15;
  const int s_l = lane >> 4;

  for (int chunk = 0; chunk < 2; ++chunk) {
    __syncthreads();   // ldsXh ready / previous main-GEMM done reading ldsA
    // ---- DFT phase: each wave does 4 channels on the WMMA units ----
    for (int cl = 0; cl < 4; ++cl) {
      const int clocal = wave*4 + cl;        // 0..15 within chunk
      const int c = chunk*16 + clocal;
      v8f accF[5];
      #pragma unroll
      for (int ct = 0; ct < 5; ++ct) accF[ct] = (v8f)(0.f);
      #pragma unroll
      for (int kt = 0; kt < 2; ++kt) {
        // build A fragment in-registers from the time series:
        // lane holds row m=m_l; half h -> sample kl = (h&7) + 8*s_l + 16*(h>>3)
        const int t0 = m_l + kt*32 + s_l*8;
        v16h af;
        #pragma unroll
        for (int h = 0; h < 8; ++h) {
          af[h]   = ldsXh[c][t0 + h];
          af[h+8] = ldsXh[c][t0 + 16 + h];
        }
        #pragma unroll
        for (int ct = 0; ct < 5; ++ct) {
          v16h bf = *(const v16h*)&WdftB[(size_t)((kt*5 + ct)*32 + lane)*16];
          accF[ct] = __builtin_amdgcn_wmma_f32_16x16x32_f16(
              false, af, false, bf, (short)0, accF[ct], false, false);
        }
      }
      // park re/im block (16 windows x 80 cols) in per-wave LDS scratch
      #pragma unroll
      for (int ct = 0; ct < 5; ++ct) {
        int j = ct*16 + m_l;
        #pragma unroll
        for (int r = 0; r < 8; ++r)
          ldsF[wave][s_l*8 + r][j] = accF[ct][r];
      }
      __syncthreads();
      // nonlinearity: bin n handled by lane n (lane 0 also does n=32)
      for (int m = 0; m < 16; ++m) {
        {
          int n = lane;
          float re = ldsF[wave][m][2*n], im = ldsF[wave][m][2*n+1];
          float mag = log1pf(sqrtf(re*re + im*im));
          float ph  = atan2f(im, re) * (1.f/PI_F);
          int kc = clocal*66 + 2*n;
          int lt = kc >> 5, kl = kc & 31;
          int id = lt*512 + a_lane(m,kl)*16 + a_h(kl);
          ldsA[id]   = (f16)mag;
          ldsA[id+1] = (f16)ph;
        }
        if (lane == 0) {
          int n = 32;
          float re = ldsF[wave][m][2*n], im = ldsF[wave][m][2*n+1];
          float mag = log1pf(sqrtf(re*re + im*im));
          float ph  = atan2f(im, re) * (1.f/PI_F);
          int kc = clocal*66 + 2*n;
          int lt = kc >> 5, kl = kc & 31;
          int id = lt*512 + a_lane(m,kl)*16 + a_h(kl);
          ldsA[id]   = (f16)mag;
          ldsA[id+1] = (f16)ph;
        }
      }
      __syncthreads();   // ldsF consumed before next channel reuses it
    }
    // ---- main GEMM phase: 33 K-tiles x 4 col-tiles per wave ----
    for (int lt = 0; lt < 33; ++lt) {
      v16h afrag = *(const v16h*)&ldsA[lt*512 + lane*16];
      int ktg = chunk*33 + lt;
      #pragma unroll
      for (int i = 0; i < 4; ++i) {
        int ct = wave*4 + i;
        v16h bfrag = *(const v16h*)&WpB[(size_t)((ktg*16 + ct)*32 + lane)*16];
        acc[i] = __builtin_amdgcn_wmma_f32_16x16x32_f16(
            false, afrag, false, bfrag, (short)0, acc[i], false, false);
      }
    }
  }
  // epilogue: tanh(acc+bp)*pi -> projA in A-fragment layout (K_out=256 -> 8 ktiles)
  const int mbase = s_l * 8;
  #pragma unroll
  for (int i = 0; i < 4; ++i) {
    int ncol = (wave*4 + i)*16 + m_l;
    float bias = bp[ncol];
    int kt = ncol >> 5, kl = ncol & 31;
    #pragma unroll
    for (int r = 0; r < 8; ++r) {
      int m = mbase + r;
      float v = tanhf(acc[i][r] + bias) * PI_F;
      projA[(size_t)((rt*8 + kt)*32 + a_lane(m,kl))*16 + a_h(kl)] = (f16)v;
    }
  }
}

// ---- generic fragment-layout WMMA GEMM with tanh epilogue ----
template<int KT, int CT, int CTW>
__global__ __launch_bounds__(128) void k_gemm(
    const f16* __restrict__ A, const f16* __restrict__ Bw,
    const float* __restrict__ bias, f16* __restrict__ D)
{
  const int tid = threadIdx.x, lane = tid & 31, wave = tid >> 5;
  const int rt = blockIdx.x;
  v8f acc[CTW];
  #pragma unroll
  for (int i = 0; i < CTW; ++i) acc[i] = (v8f)(0.f);
  for (int kt = 0; kt < KT; ++kt) {
    v16h afrag = *(const v16h*)&A[(size_t)((rt*KT + kt)*32 + lane)*16];
    #pragma unroll
    for (int i = 0; i < CTW; ++i) {
      int ct = wave*CTW + i;
      v16h bfrag = *(const v16h*)&Bw[(size_t)((kt*CT + ct)*32 + lane)*16];
      acc[i] = __builtin_amdgcn_wmma_f32_16x16x32_f16(
          false, afrag, false, bfrag, (short)0, acc[i], false, false);
    }
  }
  const int KTO = CT/2;   // output ktiles = CT*16/32
  const int mbase = (lane >> 4) * 8, nl = lane & 15;
  #pragma unroll
  for (int i = 0; i < CTW; ++i) {
    int ncol = (wave*CTW + i)*16 + nl;
    float bb = bias[ncol];
    int kt = ncol>>5, kl = ncol&31;
    #pragma unroll
    for (int r = 0; r < 8; ++r) {
      int m = mbase + r;
      float v = tanhf(acc[i][r] + bb);
      D[(size_t)((rt*KTO + kt)*32 + a_lane(m,kl))*16 + a_h(kl)] = (f16)v;
    }
  }
}

// ---- final: out[b,w] = h2 . Wout + bout; softmax-weighted sum into out[b] ----
__global__ void k_final(const f16* __restrict__ h2A, const float* __restrict__ Wout,
                        const float* __restrict__ bout, const float* __restrict__ agg,
                        const float* __restrict__ stats, float* __restrict__ out)
{
  int idx = blockIdx.x*blockDim.x + threadIdx.x;
  if (idx >= Bsz*NWIN) return;
  int b = idx / NWIN, w = idx - b*NWIN;
  int row = b*MPAD_PER_B + w;
  int rtile = row >> 4, m = row & 15;
  float sum = bout[0];
  for (int kt = 0; kt < 4; ++kt) {
    size_t base = (size_t)(rtile*4 + kt)*512;
    #pragma unroll
    for (int kl = 0; kl < 32; ++kl) {
      int k = kt*32 + kl;
      sum += (float)h2A[base + a_lane(m,kl)*16 + a_h(kl)] * Wout[k];
    }
  }
  float wt = expf(agg[w]-stats[0]) / stats[1];
  atomicAdd(&out[b], sum*wt);
}

extern "C" void kernel_launch(void* const* d_in, const int* in_sizes, int n_in,
                              void* d_out, int out_size, void* d_ws, size_t ws_size,
                              hipStream_t stream)
{
  const float* x    = (const float*)d_in[0];
  const float* Wp   = (const float*)d_in[1];
  const float* bp   = (const float*)d_in[2];
  const float* W1   = (const float*)d_in[3];
  const float* b1   = (const float*)d_in[4];
  const float* W2   = (const float*)d_in[5];
  const float* b2   = (const float*)d_in[6];
  const float* Wout = (const float*)d_in[7];
  const float* bout = (const float*)d_in[8];
  const float* agg  = (const float*)d_in[9];
  float* out = (float*)d_out;

  char* ws = (char*)d_ws;
  size_t off = 0;
  auto take = [&](size_t bytes)->char* {
    char* p = ws + off; off = (off + bytes + 255) & ~(size_t)255; return p;
  };
  f16*   WpB   = (f16*)take((size_t)66*16*512*sizeof(f16));        // ~1.08 MB
  f16*   W1B   = (f16*)take((size_t)8*16*512*sizeof(f16));
  f16*   W2B   = (f16*)take((size_t)8*8*512*sizeof(f16));
  f16*   WdftB = (f16*)take((size_t)2*5*512*sizeof(f16));          // 10 KB DFT basis
  f16*   projA = (f16*)take((size_t)ROWTILES*8*512*sizeof(f16));   // ~16 MB
  f16*   h1A   = (f16*)take((size_t)ROWTILES*8*512*sizeof(f16));   // ~16 MB
  f16*   h2A   = (f16*)take((size_t)ROWTILES*4*512*sizeof(f16));   // ~8 MB
  float* stats = (float*)take(8*sizeof(float));

  k_convW<<<(256*2112+255)/256, 256, 0, stream>>>(Wp, WpB, 256, 2112);
  k_convW<<<(256*256 +255)/256, 256, 0, stream>>>(W1, W1B, 256, 256);
  k_convW<<<(128*256 +255)/256, 256, 0, stream>>>(W2, W2B, 128, 256);
  k_dftbasis<<<(64*80+255)/256, 256, 0, stream>>>(WdftB);
  k_stats<<<1, 1024, 0, stream>>>(agg, stats, out);

  k_dft_gemm<<<ROWTILES, 128, 0, stream>>>(x, WdftB, WpB, bp, projA);
  k_gemm<8,16,4><<<ROWTILES, 128, 0, stream>>>(projA, W1B, b1, h1A);
  k_gemm<8, 8,2><<<ROWTILES, 128, 0, stream>>>(h1A,  W2B, b2, h2A);
  k_final<<<(Bsz*NWIN+255)/256, 256, 0, stream>>>(h2A, Wout, bout, agg, stats, out);
}
